// VADPerceptionTransformer_15006615734119
// MI455X (gfx1250) — matrix operand order repaired and checked by reading, not compiled, not run on hardware
//
#include <hip/hip_runtime.h>
#include <hip/hip_bf16.h>

typedef __attribute__((ext_vector_type(16))) _Float16 v16h;
typedef __attribute__((ext_vector_type(8)))  _Float16 v8h;
typedef __attribute__((ext_vector_type(8)))  float    v8f;

#define EMBED      256
#define NUM_HEADS  8
#define HEAD_DIM   32
#define NUM_POINTS 8
#define QPN        192   // 128 offset cols + 64 attn cols

// ---------------------------------------------------------------------------
// f32 -> f16 elementwise convert
// ---------------------------------------------------------------------------
__global__ void cvt_f32_f16_kernel(const float* __restrict__ in,
                                   _Float16* __restrict__ out, int n) {
  int i = blockIdx.x * blockDim.x + threadIdx.x;
  if (i < n) out[i] = (_Float16)in[i];
}

// transpose K x N (row-major) f32 -> N x K (row-major) f16 : out[n*K+k] = in[k*N+n]
__global__ void tcvt_f32_f16_kernel(const float* __restrict__ in,
                                    _Float16* __restrict__ out, int K, int N) {
  int i = blockIdx.x * blockDim.x + threadIdx.x;
  if (i >= K * N) return;
  int n = i / K;
  int k = i - n * K;
  out[i] = (_Float16)in[k * N + n];
}

// ---------------------------------------------------------------------------
// WMMA GEMM: C(MxN,f32) = A(MxK,f16,row-major) * B (given as BT: NxK,f16) +bias
// One wave computes one 16x16 tile; K stepped by 32 via v_wmma_f32_16x16x32_f16.
// M,N multiples of 16; K multiple of 32. Whole-wave uniform control flow only.
// ---------------------------------------------------------------------------
__global__ void wmma_gemm_kernel(const _Float16* __restrict__ A,
                                 const _Float16* __restrict__ BT,
                                 const float* __restrict__ bias,
                                 float* __restrict__ C,
                                 int M, int N, int K) {
  const int wid  = blockIdx.x * (blockDim.x >> 5) + (threadIdx.x >> 5);
  const int lane = threadIdx.x & 31;
  const int ntiles_n = N >> 4;
  const int ntiles   = (M >> 4) * ntiles_n;
  if (wid >= ntiles) return;                 // uniform per wave -> EXEC all 1s

  const int mt = wid / ntiles_n;
  const int nt = wid - mt * ntiles_n;
  const int half = lane >> 4;                // 0: lanes 0-15, 1: lanes 16-31
  const int l15  = lane & 15;

  const _Float16* arow = A  + (size_t)(mt * 16 + l15) * K;  // A row for this lane
  const _Float16* bcol = BT + (size_t)(nt * 16 + l15) * K;  // B column (transposed)

  v8f acc = {};
  for (int k0 = 0; k0 < K; k0 += 32) {
    // ISA 16-bit A 16x32 layout: lanes0-15 hold K0..7 (v0-3) and K16..23 (v4-7);
    // lanes16-31 hold K8..15 and K24..31.
    v8h a0 = *(const v8h*)(arow + k0 + half * 8);
    v8h a1 = *(const v8h*)(arow + k0 + half * 8 + 16);
    // B 32x16: same K order as A: lanes0-15 hold K0..7 and K16..23, lanes16-31 hold K8..15 and K24..31 (via BT).
    v8h b0 = *(const v8h*)(bcol + k0 + half * 8);
    v8h b1 = *(const v8h*)(bcol + k0 + half * 8 + 16);
    v16h a, b;
#pragma unroll
    for (int i = 0; i < 8; ++i) {
      a[i] = a0[i]; a[i + 8] = a1[i];
      b[i] = b0[i]; b[i + 8] = b1[i];
    }
    acc = __builtin_amdgcn_wmma_f32_16x16x32_f16(
        /*neg_a=*/false, a, /*neg_b=*/false, b,
        /*c_mod=*/(short)0, acc, /*reuse_a=*/false, /*reuse_b=*/false);
  }

  // C/D layout: VGPR i -> row (half*8 + i), col = lane&15 within tile.
  const int col = nt * 16 + l15;
  const float bv = bias ? bias[col] : 0.0f;
  const int row0 = mt * 16 + half * 8;
#pragma unroll
  for (int i = 0; i < 8; ++i)
    C[(size_t)(row0 + i) * N + col] = acc[i] + bv;
}

// ---------------------------------------------------------------------------
// Deformable sampling: one wave per (b, q, h); lane = head-dim channel.
// vproj: (bs, nv, 256) f32 ; qproj: (bs*nq, 192) f32 (no bias yet)
// ---------------------------------------------------------------------------
__global__ void deform_sample_kernel(const float* __restrict__ vproj,
                                     const float* __restrict__ qproj,
                                     const float* __restrict__ refpts,
                                     const float* __restrict__ b_off,
                                     const float* __restrict__ b_attn,
                                     float* __restrict__ out,
                                     int rows_q, int rows_v,
                                     const int* __restrict__ Hp,
                                     const int* __restrict__ Wp) {
  const int wid  = blockIdx.x * (blockDim.x >> 5) + (threadIdx.x >> 5);
  const int lane = threadIdx.x & 31;
  const int total = rows_q * NUM_HEADS;
  if (wid >= total) return;

  const int H  = Hp[0];
  const int W  = Wp[0];
  const int nv = H * W;
  const int bs = rows_v / nv;       // bs from value rows
  const int nq = rows_q / bs;

  const int h = wid & (NUM_HEADS - 1);
  const int r = wid >> 3;            // flat (b*nq + q)
  const int b = r / nq;

  const float* qp = qproj + (size_t)r * QPN;

  // --- attention softmax over the 8 points (wave-uniform, redundant per lane)
  float e[NUM_POINTS];
  float mx = -3.0e38f;
#pragma unroll
  for (int p = 0; p < NUM_POINTS; ++p) {
    e[p] = qp[128 + h * NUM_POINTS + p] + b_attn[h * NUM_POINTS + p];
    mx = fmaxf(mx, e[p]);
  }
  float s = 0.0f;
#pragma unroll
  for (int p = 0; p < NUM_POINTS; ++p) { e[p] = __expf(e[p] - mx); s += e[p]; }
  const float inv_s = 1.0f / s;

  // --- reference points (Z = 4); point p uses z = p & 3
  float rx[4], ry[4];
#pragma unroll
  for (int z = 0; z < 4; ++z) {
    rx[z] = refpts[((size_t)r * 4 + z) * 2 + 0];
    ry[z] = refpts[((size_t)r * 4 + z) * 2 + 1];
  }

  const float* vb = vproj + (size_t)b * nv * EMBED + h * HEAD_DIM + lane;
  const float fW = (float)W, fH = (float)H;
  float acc = 0.0f;

#pragma unroll
  for (int p = 0; p < NUM_POINTS; ++p) {
    const float ox = qp[h * 16 + p * 2 + 0] + b_off[h * 16 + p * 2 + 0];
    const float oy = qp[h * 16 + p * 2 + 1] + b_off[h * 16 + p * 2 + 1];
    const int z = p & 3;
    // x = (ref_x + ox/W)*W - 0.5 = ref_x*W + ox - 0.5
    const float x = rx[z] * fW + ox - 0.5f;
    const float y = ry[z] * fH + oy - 0.5f;
    const float x0f = floorf(x), y0f = floorf(y);
    const float fx = x - x0f, fy = y - y0f;
    const int ix0 = (int)x0f, iy0 = (int)y0f;

    float samp = 0.0f;
#pragma unroll
    for (int dy = 0; dy < 2; ++dy) {
#pragma unroll
      for (int dx = 0; dx < 2; ++dx) {
        const int xc = ix0 + dx, yc = iy0 + dy;
        const float wc = (dx ? fx : 1.0f - fx) * (dy ? fy : 1.0f - fy);
        if (xc >= 0 && xc < W && yc >= 0 && yc < H) {   // wave-uniform branch
          samp += wc * vb[(size_t)(yc * W + xc) * EMBED]; // coalesced 128B line
        }
      }
    }
    acc += (e[p] * inv_s) * samp;
  }

  out[(size_t)r * EMBED + h * HEAD_DIM + lane] = acc;
}

// ---------------------------------------------------------------------------
static inline size_t align256(size_t x) { return (x + 255) & ~(size_t)255; }

extern "C" void kernel_launch(void* const* d_in, const int* in_sizes, int n_in,
                              void* d_out, int out_size, void* d_ws, size_t ws_size,
                              hipStream_t stream) {
  const float* query  = (const float*)d_in[0];
  const float* value  = (const float*)d_in[1];
  const float* refpts = (const float*)d_in[2];
  const float* w_val  = (const float*)d_in[3];
  const float* b_val  = (const float*)d_in[4];
  const float* w_off  = (const float*)d_in[5];
  const float* b_off  = (const float*)d_in[6];
  const float* w_attn = (const float*)d_in[7];
  const float* b_attn = (const float*)d_in[8];
  const int*   Hp     = (const int*)d_in[9];
  const int*   Wp     = (const int*)d_in[10];
  float* out = (float*)d_out;

  const int rows_q = in_sizes[0] / EMBED;   // bs*nq   (=20000)
  const int rows_v = in_sizes[1] / EMBED;   // bs*nv   (=29440)

  // workspace carve-up
  char* ws = (char*)d_ws;
  size_t off = 0;
  _Float16* val16 = (_Float16*)(ws + off); off = align256(off + (size_t)rows_v * EMBED * 2);
  _Float16* qry16 = (_Float16*)(ws + off); off = align256(off + (size_t)rows_q * EMBED * 2);
  _Float16* wvalT = (_Float16*)(ws + off); off = align256(off + (size_t)EMBED * EMBED * 2);
  _Float16* wqpT  = (_Float16*)(ws + off); off = align256(off + (size_t)QPN * EMBED * 2);
  float*    vproj = (float*)(ws + off);    off = align256(off + (size_t)rows_v * EMBED * 4);
  float*    qproj = (float*)(ws + off);    off = align256(off + (size_t)rows_q * QPN * 4);

  // 1) convert inputs to f16
  {
    int n = rows_v * EMBED;
    cvt_f32_f16_kernel<<<(n + 255) / 256, 256, 0, stream>>>(value, val16, n);
    n = rows_q * EMBED;
    cvt_f32_f16_kernel<<<(n + 255) / 256, 256, 0, stream>>>(query, qry16, n);
  }
  // 2) transpose-convert weights: BT[n*K + k] = W[k*N + n]
  {
    int n = EMBED * EMBED;  // w_val 256x256 -> 256x256
    tcvt_f32_f16_kernel<<<(n + 255) / 256, 256, 0, stream>>>(w_val, wvalT, EMBED, EMBED);
    n = EMBED * 128;        // w_off 256x128 -> rows 0..127 of wqpT
    tcvt_f32_f16_kernel<<<(n + 255) / 256, 256, 0, stream>>>(w_off, wqpT, EMBED, 128);
    n = EMBED * 64;         // w_attn 256x64 -> rows 128..191 of wqpT
    tcvt_f32_f16_kernel<<<(n + 255) / 256, 256, 0, stream>>>(w_attn, wqpT + (size_t)128 * EMBED, EMBED, 64);
  }
  // 3) value projection: vproj = value @ w_val + b_val   (f16 WMMA, f32 acc)
  {
    const int waves = (rows_v / 16) * (EMBED / 16);
    const int blocks = (waves + 3) / 4;
    wmma_gemm_kernel<<<blocks, 128, 0, stream>>>(val16, wvalT, b_val, vproj,
                                                 rows_v, EMBED, EMBED);
  }
  // 4) query projection (offsets ++ attn logits), biases added in sampling
  {
    const int waves = (rows_q / 16) * (QPN / 16);
    const int blocks = (waves + 3) / 4;
    wmma_gemm_kernel<<<blocks, 128, 0, stream>>>(qry16, wqpT, nullptr, qproj,
                                                 rows_q, QPN, EMBED);
  }
  // 5) softmax + bilinear sampling + head reduction
  {
    const int waves = rows_q * NUM_HEADS;        // one wave per (b,q,h)
    const int blocks = (waves + 7) / 8;          // 8 waves / 256-thread block
    deform_sample_kernel<<<blocks, 256, 0, stream>>>(vproj, qproj, refpts,
                                                     b_off, b_attn, out,
                                                     rows_q, rows_v, Hp, Wp);
  }
}
